// Convolution_FKAConvAP_74440373174351
// MI455X (gfx1250) — matrix operand order repaired and checked by reading, not compiled, not run on hardware
//
#include <hip/hip_runtime.h>
#include <math.h>

// ---------------- problem constants (match reference) ----------------
#define BB     2
#define NN     8192
#define KNB    16      // neighbors
#define KS     16
#define KT     48      // 3*KNB k-slots
#define COUT   64
#define EPS_IN 1e-5f

// workspace layout (floats)
#define SQ_OFF 0                 // B*N squared norms of pos
#define AV_OFF (BB*NN)           // B accumulators for av_dist
#define S1_OFF (AV_OFF + BB)     // per (b): [0..15]=sum(z1), [16..31]=sumsq(z1)
#define S2_OFF (S1_OFF + BB*32)  // per (b): same for z2
#define POSH_OFF (S2_OFF + BB*32) // B*3*N f16 copy of pos (TDM source), 24576 floats
#define ACC_CNT (BB + BB*32 + BB*32)

// knn LDS panel: pos f16 [3][N] (48KB) + sq f32 [N] (32KB)
#define LDS_SQ_BYTE (3 * NN * 2)
#define LDS_TOTAL   (LDS_SQ_BYTE + NN * 4)

typedef __attribute__((ext_vector_type(16))) _Float16 v16h;
typedef __attribute__((ext_vector_type(8)))  float    v8f;
typedef __attribute__((ext_vector_type(4)))  unsigned int ui4;
typedef __attribute__((ext_vector_type(8)))  int      i8v;
typedef __attribute__((ext_vector_type(4)))  int      i4v;

__device__ inline v16h vzero16() {
  v16h r;
#pragma unroll
  for (int i = 0; i < 16; ++i) r[i] = (_Float16)0.0f;
  return r;
}
__device__ inline v8f vzero8() {
  v8f r;
#pragma unroll
  for (int i = 0; i < 8; ++i) r[i] = 0.0f;
  return r;
}

// wave32 cross-lane fetch (full-EXEC call sites only)
__device__ inline float xshfl(float v, int srcLane) {
  return __int_as_float(__builtin_amdgcn_ds_bpermute(srcLane << 2, __float_as_int(v)));
}

// ---- Tensor Data Mover: 1-row flat copy global -> LDS (6-arg builtin form) ----
// D# group0: count=1, lds_addr[63:32], global_addr[120:64], type=2[127:126]
// D# group1: data_size[17:16], tensor_dim0[79:48], tensor_dim1[111:80]=1,
//            tile_dim0[127:112], tile_dim1[143:128]=1, dim0_stride[207:160]
__device__ inline void tdm_load_1d(unsigned lds_byte, const void* gptr,
                                   unsigned elems, int dsz_code) {
  const unsigned long long ga = (unsigned long long)(size_t)gptr;
  ui4 g0;
  g0.x = 1u;                                              // count=1, user mode
  g0.y = lds_byte;                                        // lds_addr
  g0.z = (unsigned)(ga & 0xFFFFFFFFu);                    // global_addr[31:0]
  g0.w = (unsigned)((ga >> 32) & 0x1FFFFFFu) | (2u << 30); // addr[56:32] | type=2
  i8v g1;
  g1[0] = (int)((unsigned)dsz_code << 16);                // mask=0, data_size
  g1[1] = (int)((elems & 0xFFFFu) << 16);                 // tensor_dim0[15:0]
  g1[2] = (int)((elems >> 16) | (1u << 16));              // tensor_dim0[31:16], tdim1=1
  g1[3] = (int)((elems & 0xFFFFu) << 16);                 // tile_dim0 = elems
  g1[4] = 1;                                              // tile_dim1=1, tile_dim2=0
  g1[5] = (int)elems;                                     // dim0_stride[31:0]
  g1[6] = (int)((unsigned long long)elems >> 32);         // dim0_stride[47:32]
  g1[7] = 0;
  const i4v z4 = {0, 0, 0, 0};
  const i8v z8 = {0, 0, 0, 0, 0, 0, 0, 0};
  __builtin_amdgcn_tensor_load_to_lds(g0, g1, z4, z4, z8, 0);
}

// Stage a 16x32 f32 weight matrix (row-major [16][32]) into CDNA5 WMMA
// 16-bit A operand, in this file's contiguous k order (16x32 MxK, wave32):
//   lane<16 : K=ei       (M = lane)
//   lane>=16: K=ei+16    (M = lane-16)
__device__ inline v16h stage_A16x32(const float* W, int lane) {
  const int m  = lane & 15;
  const int hi = lane >> 4;
  v16h a;
#pragma unroll
  for (int ei = 0; ei < 16; ++ei) {
    const int k = ei + (hi ? 16 : 0);
    a[ei] = (_Float16)W[m * 32 + k];
  }
  return a;
}

struct Params {
  const float* x;
  const float* pos;
  const float* sup;
  const int*   nbr;
  const float* fc1;
  const float* fc2;
  const float* fc3;
  const float* bn1w; const float* bn1b;
  const float* bn2w; const float* bn2b;
  const float* cv;
  const float* alpha; const float* beta;
  float* ws;
  float* out;
};

// ---------------- kernel 0: squared norms + f16 pos copy + zero accumulators ----------------
__global__ __launch_bounds__(256) void prep_kernel(const float* pos, float* ws) {
  const int i = blockIdx.x * 256 + threadIdx.x;
  if (i < BB * NN) {
    const int b = i / NN, n = i % NN;
    _Float16* ph = (_Float16*)(ws + POSH_OFF);
    float s = 0.f;
#pragma unroll
    for (int c = 0; c < 3; ++c) {
      const float v = pos[(b * 3 + c) * NN + n];
      ph[(b * 3 + c) * NN + n] = (_Float16)v;
      s += v * v;
    }
    ws[SQ_OFF + i] = s;
  }
  if (i < ACC_CNT) ws[AV_OFF + i] = 0.f;
}

// ---------------- kernel 1: N^2 knn (top-2) via WMMA f16 16x16x32 ----------------
// Whole batch column panel staged in LDS once per workgroup via the Tensor Data
// Mover; the 512-tile loop then runs entirely out of LDS.
__global__ __launch_bounds__(256) void knn_kernel(float* ws) {
  extern __shared__ char smemraw[];
  const int lane = threadIdx.x & 31;
  const int wid  = (blockIdx.x * 256 + threadIdx.x) >> 5;   // 0 .. B*N/16-1
  const int b    = wid / (NN / 16);                         // uniform per block
  const int rb   = (wid % (NN / 16)) * 16;
  float* av = ws + AV_OFF;

  // TDM: stage pos_h[b] (3*N f16) and sq[b] (N f32) into LDS (wave 0 only)
  const unsigned lds_base = (unsigned)(size_t)smemraw;
  if (threadIdx.x < 32) {
    const _Float16* phg = (const _Float16*)(ws + POSH_OFF) + b * 3 * NN;
    const float*    sqg = ws + SQ_OFF + b * NN;
    tdm_load_1d(lds_base,               phg, 3 * NN, 1);    // 2-byte elems
    tdm_load_1d(lds_base + LDS_SQ_BYTE, sqg, NN,     2);    // 4-byte elems
    __builtin_amdgcn_s_wait_tensorcnt(0);
  }
  __syncthreads();
  const _Float16* lph = (const _Float16*)smemraw;
  const float*    lsq = (const float*)(smemraw + LDS_SQ_BYTE);

  v16h A = vzero16();
  {
    const int m = lane & 15;
    const float sel = (lane < 16) ? 1.f : 0.f;   // lanes>=16 carry K=8..15,24..31 -> zero
    A[0] = (_Float16)((float)lph[0 * NN + rb + m] * sel);
    A[1] = (_Float16)((float)lph[1 * NN + rb + m] * sel);
    A[2] = (_Float16)((float)lph[2 * NN + rb + m] * sel);
  }
  const int r0 = rb + ((lane >> 4) << 3);        // first of this lane's 8 rows
  float sqR[8];
#pragma unroll
  for (int r = 0; r < 8; ++r) sqR[r] = lsq[r0 + r];

  float m1[8], m2[8];
#pragma unroll
  for (int r = 0; r < 8; ++r) { m1[r] = 3.402823e38f; m2[r] = 3.402823e38f; }

  for (int cb = 0; cb < NN; cb += 16) {
    const int col = cb + (lane & 15);
    const bool lo = (lane < 16);
    v16h Bv = vzero16();
    Bv[0] = lo ? lph[0 * NN + col] : (_Float16)0.f;
    Bv[1] = lo ? lph[1 * NN + col] : (_Float16)0.f;
    Bv[2] = lo ? lph[2 * NN + col] : (_Float16)0.f;
    const float sqc = lsq[col];

    v8f cz = vzero8();
    v8f G = __builtin_amdgcn_wmma_f32_16x16x32_f16(false, A, false, Bv,
                                                   (short)0, cz, false, false);
#pragma unroll
    for (int r = 0; r < 8; ++r) {
      float d2 = sqR[r] + sqc - 2.f * G[r];
      const int row = r0 + r;
      d2 = (row == col) ? 0.f : fmaxf(d2, 0.f);
      const float nm1 = fminf(m1[r], d2);
      const float nm2 = fminf(m2[r], fmaxf(m1[r], d2));
      m1[r] = nm1; m2[r] = nm2;
    }
  }
  // merge top-2 across the 16 lanes sharing each row (xor within half-wave)
#pragma unroll
  for (int mask = 1; mask <= 8; mask <<= 1) {
#pragma unroll
    for (int r = 0; r < 8; ++r) {
      const float o1 = xshfl(m1[r], lane ^ mask);
      const float o2 = xshfl(m2[r], lane ^ mask);
      const float nm1 = fminf(m1[r], o1);
      const float nm2 = fminf(fmaxf(m1[r], o1), fminf(m2[r], o2));
      m1[r] = nm1; m2[r] = nm2;
    }
  }
  float s = 0.f;
#pragma unroll
  for (int r = 0; r < 8; ++r) s += sqrtf(m1[r]) + sqrtf(m2[r]);
  s += xshfl(s, lane ^ 16);                       // combine the two row halves
  if (lane == 0) atomicAdd(&av[b], s);
}

// ---------------- per-point pipeline: STAGE 1 = z1 stats, 2 = z2 stats, 3 = final ----------------
// One wave per point. Lane owns k-slot colA = lane (groups 0/1) and, for lanes<16,
// colB = lane+32 (group 2). 16-col tile t of the 48 k-slots feeds one WMMA.
template <int STAGE>
__global__ __launch_bounds__(256) void point_kernel(Params p) {
  __shared__ float smem[(STAGE == 3) ? (COUT * 5 * KS) : 32];
  if constexpr (STAGE == 3) {
    for (int i = threadIdx.x; i < COUT * 5 * KS; i += 256) smem[i] = p.cv[i];
  } else {
    if (threadIdx.x < 32) smem[threadIdx.x] = 0.f;
  }
  __syncthreads();

  const int lane = threadIdx.x & 31;
  const int wid  = (blockIdx.x * 256 + threadIdx.x) >> 5;   // 0 .. B*N-1
  const int b = wid / NN, n = wid % NN;
  const int j  = lane & 15;
  const int hi = lane >> 4;

  const float alpha = p.alpha[0], beta = p.beta[0];
  const float ad = p.ws[AV_OFF + b] * (1.f / (2.f * (float)NN));

  // gather neighbor j
  const int nb = p.nbr[(b * NN + n) * KNB + j];
  float pg[3], xg[3], sp[3];
#pragma unroll
  for (int c = 0; c < 3; ++c) {
    pg[c] = p.pos[(b * 3 + c) * NN + nb];
    xg[c] = p.x  [(b * 3 + c) * NN + nb];
    sp[c] = p.sup[(b * 3 + c) * NN + n];
  }
  // colA: group 0 (lanes<16) or 1 (lanes>=16); colB: group 2
  const float sgnA = hi ? -1.f : 0.f;
  float ptsA[3], ptsB[3];
#pragma unroll
  for (int c = 0; c < 3; ++c) {
    const float base = pg[c] - sp[c];
    ptsA[c] = base + sgnA * xg[c] * ad;
    ptsB[c] = base + xg[c] * ad;
  }
  const float dA = sqrtf(ptsA[0]*ptsA[0] + ptsA[1]*ptsA[1] + ptsA[2]*ptsA[2]);
  const float dB = sqrtf(ptsB[0]*ptsB[0] + ptsB[1]*ptsB[1] + ptsB[2]*ptsB[2]);
  float dwA = 1.f / (1.f + expf(alpha * dA - beta));
  float dwB = 1.f / (1.f + expf(alpha * dB - beta));
  float ssum = dwA + (hi ? 0.f : dwB);
#pragma unroll
  for (int mask = 1; mask <= 16; mask <<= 1) ssum += xshfl(ssum, lane ^ mask);
  ssum = ssum + ((ssum == 0.f) ? 1.f : 0.f) + 1e-6f;
  const float dscale = (float)KT / ssum;
  dwA *= dscale; dwB *= dscale;

  // z1 = fc1[16x3] * pts  (per owned column)
  float z1A[16], z1B[16];
#pragma unroll
  for (int o = 0; o < 16; ++o) {
    const float w0 = p.fc1[o*3+0], w1 = p.fc1[o*3+1], w2 = p.fc1[o*3+2];
    z1A[o] = w0*ptsA[0] + w1*ptsA[1] + w2*ptsA[2];
    z1B[o] = w0*ptsB[0] + w1*ptsB[1] + w2*ptsB[2];
  }

  const float invCnt = 1.f / ((float)NN * (float)KT);

  if constexpr (STAGE == 1) {
    float ps[16], pq[16];
#pragma unroll
    for (int o = 0; o < 16; ++o) {
      const float a = z1A[o], bv = hi ? 0.f : z1B[o];
      ps[o] = a + bv; pq[o] = a*a + bv*bv;
    }
#pragma unroll
    for (int mask = 1; mask <= 16; mask <<= 1) {
#pragma unroll
      for (int o = 0; o < 16; ++o) {
        ps[o] += xshfl(ps[o], lane ^ mask);
        pq[o] += xshfl(pq[o], lane ^ mask);
      }
    }
    if (lane == 0) {
#pragma unroll
      for (int o = 0; o < 16; ++o) {
        atomicAdd(&smem[o],      ps[o]);
        atomicAdd(&smem[16 + o], pq[o]);
      }
    }
    __syncthreads();
    if (threadIdx.x < 32) atomicAdd(&p.ws[S1_OFF + b * 32 + threadIdx.x], smem[threadIdx.x]);
    return;
  } else {
    // ---- mat1 = relu(inorm(z1)), mp1 = max_k(mat1*dw) ----
    const float* s1 = p.ws + S1_OFF + b * 32;
    float mat1A[16], mat1B[16], mp1[16];
#pragma unroll
    for (int o = 0; o < 16; ++o) {
      const float m  = s1[o] * invCnt;
      const float v  = s1[16 + o] * invCnt - m * m;
      const float sc = rsqrtf(v + EPS_IN) * p.bn1w[o];
      const float of = p.bn1b[o] - m * sc;
      mat1A[o] = fmaxf(z1A[o] * sc + of, 0.f);
      mat1B[o] = fmaxf(z1B[o] * sc + of, 0.f);
      mp1[o]   = fmaxf(mat1A[o] * dwA, hi ? -3.4e38f : mat1B[o] * dwB);
    }
#pragma unroll
    for (int mask = 1; mask <= 16; mask <<= 1)
#pragma unroll
      for (int o = 0; o < 16; ++o) mp1[o] = fmaxf(mp1[o], xshfl(mp1[o], lane ^ mask));

    // ---- z2 = fc2[16x32] x [mat1 ; mp1][32x48] via 3 WMMAs ----
    const v16h Af2 = stage_A16x32(p.fc2, lane);
    float m1X[16];
#pragma unroll
    for (int o = 0; o < 16; ++o) m1X[o] = xshfl(mat1A[o], lane ^ 16); // cols 16..31
    v8f z2t[3];
#pragma unroll
    for (int t = 0; t < 3; ++t) {
      v16h Bv;
#pragma unroll
      for (int ei = 0; ei < 16; ++ei) {
        const float lowv = (t == 0) ? mat1A[ei] : ((t == 1) ? m1X[ei] : mat1B[ei]);
        Bv[ei] = (_Float16)(hi ? mp1[ei] : lowv);  // lanes>=16 carry K=16..31 = mp1
      }
      v8f cz = vzero8();
      z2t[t] = __builtin_amdgcn_wmma_f32_16x16x32_f16(false, Af2, false, Bv,
                                                      (short)0, cz, false, false);
    }

    if constexpr (STAGE == 2) {
      float s8[8], q8[8];
#pragma unroll
      for (int r = 0; r < 8; ++r) {
        s8[r] = z2t[0][r] + z2t[1][r] + z2t[2][r];
        q8[r] = z2t[0][r]*z2t[0][r] + z2t[1][r]*z2t[1][r] + z2t[2][r]*z2t[2][r];
      }
#pragma unroll
      for (int mask = 1; mask <= 8; mask <<= 1) {
#pragma unroll
        for (int r = 0; r < 8; ++r) {
          s8[r] += xshfl(s8[r], lane ^ mask);
          q8[r] += xshfl(q8[r], lane ^ mask);
        }
      }
      if (j == 0) {                 // lanes 0 and 16 hold the half totals
        const int ob = hi * 8;
#pragma unroll
        for (int r = 0; r < 8; ++r) {
          atomicAdd(&smem[ob + r],      s8[r]);
          atomicAdd(&smem[16 + ob + r], q8[r]);
        }
      }
      __syncthreads();
      if (threadIdx.x < 32) atomicAdd(&p.ws[S2_OFF + b * 32 + threadIdx.x], smem[threadIdx.x]);
      return;
    } else {
      // ---- STAGE 3: mat2, mp2, fc3 WMMA, mat3, feat, out ----
      const float* s2 = p.ws + S2_OFF + b * 32;
      const int ob = hi * 8;                  // this half's row base in C layout
      float mat2[3][8];
#pragma unroll
      for (int r = 0; r < 8; ++r) {
        const int o = ob + r;
        const float m  = s2[o] * invCnt;
        const float v  = s2[16 + o] * invCnt - m * m;
        const float sc = rsqrtf(v + EPS_IN) * p.bn2w[o];
        const float of = p.bn2b[o] - m * sc;
#pragma unroll
        for (int t = 0; t < 3; ++t) mat2[t][r] = fmaxf(z2t[t][r] * sc + of, 0.f);
      }
      const float dw0 = xshfl(dwA, j);        // dw of this C-column in tile 0/1/2
      const float dw1 = xshfl(dwA, 16 + j);
      const float dw2 = xshfl(dwB, j);

      float mp2h[8];
#pragma unroll
      for (int r = 0; r < 8; ++r)
        mp2h[r] = fmaxf(fmaxf(mat2[0][r]*dw0, mat2[1][r]*dw1), mat2[2][r]*dw2);
#pragma unroll
      for (int mask = 1; mask <= 8; mask <<= 1)
#pragma unroll
        for (int r = 0; r < 8; ++r) mp2h[r] = fmaxf(mp2h[r], xshfl(mp2h[r], lane ^ mask));
      float mp2o[8];
#pragma unroll
      for (int r = 0; r < 8; ++r) mp2o[r] = xshfl(mp2h[r], lane ^ 16);
      float mp2f[16];
#pragma unroll
      for (int e = 0; e < 8; ++e) {
        mp2f[e]     = hi ? mp2o[e] : mp2h[e];
        mp2f[8 + e] = hi ? mp2h[e] : mp2o[e];
      }
      float matX[3][8];                        // other half's rows, same column
#pragma unroll
      for (int t = 0; t < 3; ++t)
#pragma unroll
        for (int r = 0; r < 8; ++r) matX[t][r] = xshfl(mat2[t][r], lane ^ 16);

      const v16h Af3 = stage_A16x32(p.fc3, lane);
      v8f z3t[3];
#pragma unroll
      for (int t = 0; t < 3; ++t) {
        v16h Bv;
#pragma unroll
        for (int ei = 0; ei < 16; ++ei) {
          const float lowv = (ei < 8) ? mat2[t][ei] : matX[t][ei - 8];
          Bv[ei] = (_Float16)(hi ? mp2f[ei] : lowv);
        }
        v8f cz = vzero8();
        z3t[t] = __builtin_amdgcn_wmma_f32_16x16x32_f16(false, Af3, false, Bv,
                                                        (short)0, cz, false, false);
      }
      // mat3 = relu(z3)*dw ; feat partials (xfull[c][k]=x_g[c][k%16] for c<3)
      float S[8], p3[8], p4[8];
#pragma unroll
      for (int r = 0; r < 8; ++r) {
        const float a0 = fmaxf(z3t[0][r], 0.f) * dw0;
        const float a1 = fmaxf(z3t[1][r], 0.f) * dw1;
        const float a2 = fmaxf(z3t[2][r], 0.f) * dw2;
        S[r] = a0 + a1 + a2; p3[r] = a1; p4[r] = a2;
      }
      float fp[5][8];
#pragma unroll
      for (int r = 0; r < 8; ++r) {
        fp[0][r] = xg[0] * S[r];
        fp[1][r] = xg[1] * S[r];
        fp[2][r] = xg[2] * S[r];
        fp[3][r] = p3[r];
        fp[4][r] = p4[r];
      }
#pragma unroll
      for (int mask = 1; mask <= 8; mask <<= 1)
#pragma unroll
        for (int c = 0; c < 5; ++c)
#pragma unroll
          for (int r = 0; r < 8; ++r) fp[c][r] += xshfl(fp[c][r], lane ^ mask);

      // out[o2] = sum_{c,o} cv[o2][c][o] * feat[c][o]  (cv staged in LDS)
      float po[4];
#pragma unroll
      for (int jj = 0; jj < 4; ++jj) {
        const int o2 = j + jj * 16;
        const float* cvb = &smem[o2 * 80];     // [5][16]
        float acc = 0.f;
#pragma unroll
        for (int c = 0; c < 5; ++c)
#pragma unroll
          for (int r = 0; r < 8; ++r) acc += cvb[c * 16 + ob + r] * fp[c][r];
        po[jj] = acc;
      }
#pragma unroll
      for (int jj = 0; jj < 4; ++jj) po[jj] += xshfl(po[jj], lane ^ 16);
      if (!hi) {
#pragma unroll
        for (int jj = 0; jj < 4; ++jj) {
          const int o2 = j + jj * 16;
          p.out[((b * COUT) + o2) * NN + n] = po[jj];
        }
      }
    }
  }
}

extern "C" void kernel_launch(void* const* d_in, const int* in_sizes, int n_in,
                              void* d_out, int out_size, void* d_ws, size_t ws_size,
                              hipStream_t stream) {
  Params P;
  P.x     = (const float*)d_in[0];
  P.pos   = (const float*)d_in[1];
  P.sup   = (const float*)d_in[2];
  P.nbr   = (const int*)  d_in[3];
  P.fc1   = (const float*)d_in[4];
  P.fc2   = (const float*)d_in[5];
  P.fc3   = (const float*)d_in[6];
  P.bn1w  = (const float*)d_in[7];
  P.bn1b  = (const float*)d_in[8];
  P.bn2w  = (const float*)d_in[9];
  P.bn2b  = (const float*)d_in[10];
  P.cv    = (const float*)d_in[11];
  P.alpha = (const float*)d_in[12];
  P.beta  = (const float*)d_in[13];
  P.ws    = (float*)d_ws;
  P.out   = (float*)d_out;

  prep_kernel<<<(BB * NN + 255) / 256, 256, 0, stream>>>(P.pos, P.ws);
  knn_kernel <<<(BB * NN / 16) / 8, 256, LDS_TOTAL, stream>>>(P.ws);
  point_kernel<1><<<BB * NN / 8, 256, 0, stream>>>(P);
  point_kernel<2><<<BB * NN / 8, 256, 0, stream>>>(P);
  point_kernel<3><<<BB * NN / 8, 256, 0, stream>>>(P);
}
